// GatheringLossDim_7739531067608
// MI455X (gfx1250) — matrix operand order repaired and checked by reading, not compiled, not run on hardware
//
#include <hip/hip_runtime.h>

// ---------------------------------------------------------------------------
// queries (8,8,1024,512) f32, items (2048,512) f32 -> out (8,1024) f32
// softmax is monotonic => argmax over logits; loss recomputed exactly in f32.
// Compute-bound (137 GFLOP vs ~0.3 GB traffic) => bf16 WMMA 16x16x32.
// ---------------------------------------------------------------------------
#define NN 512        // feature dim
#define MM 2048       // number of items
#define RR 65536      // B*K*T rows
#define ROWS_WG 128   // query rows per workgroup (8 waves x 16 rows)
#define BSTRIDE 520   // padded bf16 row stride in LDS (1040 B -> 4-bank skew)
#define TROWS 32      // items rows per staged tile (two 16-col WMMA tiles)
#define NTILES (MM / TROWS)
#define BUFBYTES (TROWS * BSTRIDE * 2)   // 33280 B per items tile buffer

typedef __attribute__((ext_vector_type(16))) __bf16 v16bf;
typedef __attribute__((ext_vector_type(8)))  float  v8f;
typedef __attribute__((ext_vector_type(4)))  unsigned int v4u;
typedef __attribute__((ext_vector_type(8)))  int v8i;
typedef __attribute__((ext_vector_type(4)))  int v4i;

#if defined(__has_builtin)
#  if __has_builtin(__builtin_amdgcn_tensor_load_to_lds)
#    define HAVE_TDM 1
#  endif
#endif

// Load bf16 [p..p+7] and [p+16..p+23] from LDS into a WMMA operand register block.
__device__ __forceinline__ v16bf load_bf16x16(const __bf16* p) {
    struct alignas(16) P { uint4 a, b; } t;
    t.a = *(const uint4*)(p);
    t.b = *(const uint4*)(p + 16);
    return __builtin_bit_cast(v16bf, t);
}

#if defined(HAVE_TDM)
// Tensor Data Mover: DMA one 32x512 bf16 tile (rows of a 2048x512 tensor) into
// LDS, inserting 16 B of padding after every 1024 B row (=> BSTRIDE 520).
// D# per CDNA5 ISA 8.3/8.4: group0 {count=1, lds_addr, global_addr, type=2},
// group1 {data_size=2B, pad cfg, tensor/tile dims, dim0 stride}, groups 2/3 = 0 (2D).
__device__ __forceinline__ void tdm_load_tile(const __bf16* gsrc, unsigned lds_off) {
    unsigned long long ga = (unsigned long long)(size_t)gsrc;
    v4u g0;
    g0[0] = 1u;                                          // count=1 (valid), user mode
    g0[1] = lds_off;                                     // lds_addr (bytes)
    g0[2] = (unsigned)(ga & 0xffffffffu);                // global_addr[31:0]
    g0[3] = (unsigned)((ga >> 32) & 0x1ffffffu)          // global_addr[56:32]
          | (2u << 30);                                  // type = 2 ("image")
    v8i g1;
    g1[0] = (1 << 16)      // data_size: 2 bytes
          | (1 << 20)      // pad_enable
          | (7 << 22)      // pad_interval: 256 DWORDs (one 1024 B row)
          | (3 << 25);     // pad_amount: 4 DWORDs (16 B)
    g1[1] = (NN & 0xffff) << 16;                 // tensor_dim0 = 512 (bits 63:48)
    g1[2] = (MM & 0xffff) << 16;                 // tensor_dim1 = 2048 (bits 95:80)
    g1[3] = (NN & 0xffff) << 16;                 // tile_dim0 = 512 (bits 127:112)
    g1[4] = TROWS;                               // tile_dim1 = 32 rows; tile_dim2 = 0
    g1[5] = NN;                                  // tensor_dim0_stride = 512 elements
    g1[6] = 0;
    g1[7] = 0;
    v4i z4 = {0, 0, 0, 0};
#if __clang_major__ >= 23
    v8i z8 = {0, 0, 0, 0, 0, 0, 0, 0};
    __builtin_amdgcn_tensor_load_to_lds(g0, g1, z4, z4, z8, 0);
#else
    __builtin_amdgcn_tensor_load_to_lds(g0, g1, z4, z4, 0);
#endif
}
#endif

// ---------------------------------------------------------------------------
// Phase 0: items f32 -> bf16 (one pass)
// ---------------------------------------------------------------------------
__global__ __launch_bounds__(256) void items_to_bf16(const float* __restrict__ items,
                                                     unsigned int* __restrict__ out) {
    int i = blockIdx.x * blockDim.x + threadIdx.x;   // over MM*NN/2 pairs
    float2 f = ((const float2*)items)[i];
    __bf16 h0 = (__bf16)f.x;
    __bf16 h1 = (__bf16)f.y;
    unsigned int u0 = __builtin_bit_cast(unsigned short, h0);
    unsigned int u1 = __builtin_bit_cast(unsigned short, h1);
    out[i] = u0 | (u1 << 16);
}

// ---------------------------------------------------------------------------
// Phase 1: fused GEMM + running argmax + exact f32 loss per row.
// Each wave: 16-row stripe, two independent accumulator chains (32 items/iter).
// ---------------------------------------------------------------------------
__global__ __launch_bounds__(256) void argmax_wmma(const float* __restrict__ q,
                                                   const float* __restrict__ items,
                                                   const __bf16* __restrict__ itemsbf,
                                                   float* __restrict__ lossws) {
    __shared__ __bf16 bs[2][TROWS * BSTRIDE];   // double-buffered items tiles (~65 KB)
    __shared__ int    amax[ROWS_WG];
    __shared__ float  lpart[256];

    const int tid   = threadIdx.x;
    const int lane  = tid & 31;
    const int w     = tid >> 5;          // wave id 0..7
    const int l16   = lane & 15;
    const int khalf = lane >> 4;         // 0/1 half of wave32
    const int rowbase = blockIdx.x * ROWS_WG;
    const int mrow    = rowbase + w * 16 + l16;   // this lane's query row (A-matrix M)

    // ---- A fragments: 16 chunks of K=32 (ISA 16-bit A layout), 128 VGPRs ----
    v16bf afrag[16];
    {
        const float* qrow = q + (size_t)mrow * NN;
        #pragma unroll
        for (int j = 0; j < 16; ++j) {
            const int kb = j * 32 + 8 * khalf;
            float4 f0 = *(const float4*)(qrow + kb);
            float4 f1 = *(const float4*)(qrow + kb + 4);
            float4 f2 = *(const float4*)(qrow + kb + 16);
            float4 f3 = *(const float4*)(qrow + kb + 20);
            v16bf a;
            a[0]  = (__bf16)f0.x; a[1]  = (__bf16)f0.y; a[2]  = (__bf16)f0.z; a[3]  = (__bf16)f0.w;
            a[4]  = (__bf16)f1.x; a[5]  = (__bf16)f1.y; a[6]  = (__bf16)f1.z; a[7]  = (__bf16)f1.w;
            a[8]  = (__bf16)f2.x; a[9]  = (__bf16)f2.y; a[10] = (__bf16)f2.z; a[11] = (__bf16)f2.w;
            a[12] = (__bf16)f3.x; a[13] = (__bf16)f3.y; a[14] = (__bf16)f3.z; a[15] = (__bf16)f3.w;
            afrag[j] = a;
        }
    }

    float best[8];
    int   bidx[8];
    #pragma unroll
    for (int i = 0; i < 8; ++i) { best[i] = -3.4e38f; bidx[i] = 0; }

#if defined(HAVE_TDM)
    const unsigned bs_base = (unsigned)(size_t)(void*)&bs[0][0];
    // Prologue: DMA tile 0 into buffer 0.
    if (w == 0) {
        tdm_load_tile(itemsbf, bs_base);
        __builtin_amdgcn_s_wait_tensorcnt(0);
    }
    __syncthreads();
#endif

    for (int mc = 0; mc < NTILES; ++mc) {
        const int cur = mc & 1;

#if defined(HAVE_TDM)
        // Kick DMA of next tile into the other buffer; overlaps with WMMA below.
        if (w == 0 && mc + 1 < NTILES)
            tdm_load_tile(itemsbf + (size_t)(mc + 1) * TROWS * NN,
                          bs_base + (unsigned)((1 - cur) * BUFBYTES));
#else
        // Fallback: cooperative fill (32 rows x 256 dwords, 8 thr/row x 8 uint4).
        {
            const int tr = tid >> 3;
            const int tc = tid & 7;
            const uint4* src = (const uint4*)(itemsbf + (size_t)(mc * TROWS + tr) * NN) + tc * 8;
            uint4* dst = (uint4*)(&bs[cur][0] + tr * BSTRIDE) + tc * 8;
            #pragma unroll
            for (int i = 0; i < 8; ++i) dst[i] = src[i];
        }
        __syncthreads();
#endif

        // B layout (32x16 16-bit): lane = N col, khalf selects K 8..15/24..31;
        // fragment = bs[n][32*j + 8*khalf + 0..7] and bs[n][32*j + 16 + 8*khalf + 0..7].
        // Two column tiles -> two independent accumulator chains c0/c1.
        const __bf16* brow0 = &bs[cur][0] + l16 * BSTRIDE + khalf * 8;
        const __bf16* brow1 = brow0 + 16 * BSTRIDE;
        v8f c0 = {};
        v8f c1 = {};
        #pragma unroll
        for (int j = 0; j < 16; ++j) {
            v16bf b0 = load_bf16x16(brow0 + j * 32);
            v16bf b1 = load_bf16x16(brow1 + j * 32);
            c0 = __builtin_amdgcn_wmma_f32_16x16x32_bf16(
                    false, afrag[j], false, b0, (short)0, c0, false, false);
            c1 = __builtin_amdgcn_wmma_f32_16x16x32_bf16(
                    false, afrag[j], false, b1, (short)0, c1, false, false);
        }

        // C layout: col = lane&15, row(i) = i + 8*khalf. Running argmax per lane.
        const int gcol0 = mc * TROWS + l16;
        #pragma unroll
        for (int i = 0; i < 8; ++i)
            if (c0[i] > best[i]) { best[i] = c0[i]; bidx[i] = gcol0; }
        #pragma unroll
        for (int i = 0; i < 8; ++i)
            if (c1[i] > best[i]) { best[i] = c1[i]; bidx[i] = gcol0 + 16; }

#if defined(HAVE_TDM)
        if (w == 0 && mc + 1 < NTILES)
            __builtin_amdgcn_s_wait_tensorcnt(0);   // next buffer landed
        __syncthreads();                            // publish; also retires reads of cur
#else
        __syncthreads();
#endif
    }

    // ---- Cross-lane argmax reduction within each 16-lane half (wave32) ----
    #pragma unroll
    for (int off = 1; off < 16; off <<= 1) {
        #pragma unroll
        for (int i = 0; i < 8; ++i) {
            float ov = __shfl_xor(best[i], off, 32);
            int   oi = __shfl_xor(bidx[i], off, 32);
            if (ov > best[i] || (ov == best[i] && oi < bidx[i])) {
                best[i] = ov; bidx[i] = oi;
            }
        }
    }
    if (l16 == 0) {
        #pragma unroll
        for (int i = 0; i < 8; ++i)
            amax[w * 16 + khalf * 8 + i] = bidx[i];
    }
    __syncthreads();

    // ---- Exact f32 loss: 2 threads per row, 256 floats each ----
    {
        const int rr   = tid >> 1;
        const int h    = tid & 1;
        const int grow = rowbase + rr;
        const int idx  = amax[rr];
        const float4* qp = (const float4*)(q + (size_t)grow * NN) + h * 64;
        const float4* ip = (const float4*)(items + (size_t)idx * NN) + h * 64;
        float s = 0.f;
        #pragma unroll 8
        for (int n = 0; n < 64; ++n) {
            float4 a = qp[n], b = ip[n];
            float dx = a.x - b.x, dy = a.y - b.y, dz = a.z - b.z, dw = a.w - b.w;
            s += dx * dx + dy * dy + dz * dz + dw * dw;
        }
        lpart[tid] = s;
    }
    __syncthreads();
    if (tid < ROWS_WG)
        lossws[rowbase + tid] = lpart[2 * tid] + lpart[2 * tid + 1];
}

// ---------------------------------------------------------------------------
// Phase 2: out[b,t] = sum_k loss[b,k,t]  (deterministic, no atomics)
// ---------------------------------------------------------------------------
__global__ __launch_bounds__(256) void reduce_over_k(const float* __restrict__ lossws,
                                                     float* __restrict__ out) {
    int g = blockIdx.x * blockDim.x + threadIdx.x;   // 0..8191
    int b = g >> 10;
    int t = g & 1023;
    float s = 0.f;
    #pragma unroll
    for (int k = 0; k < 8; ++k)
        s += lossws[((b * 8 + k) << 10) + t];
    out[g] = s;
}

// ---------------------------------------------------------------------------
extern "C" void kernel_launch(void* const* d_in, const int* in_sizes, int n_in,
                              void* d_out, int out_size, void* d_ws, size_t ws_size,
                              hipStream_t stream) {
    const float* q     = (const float*)d_in[0];   // (8,8,1024,512) f32
    const float* items = (const float*)d_in[1];   // (2048,512) f32

    __bf16* itemsbf = (__bf16*)d_ws;                                    // 2 MB
    float*  lossws  = (float*)((char*)d_ws + (size_t)MM * NN * 2);      // 256 KB

    items_to_bf16<<<(MM * NN / 2) / 256, 256, 0, stream>>>(items, (unsigned int*)itemsbf);
    argmax_wmma<<<RR / ROWS_WG, 256, 0, stream>>>(q, items, itemsbf, lossws);
    reduce_over_k<<<8192 / 256, 256, 0, stream>>>(lossws, (float*)d_out);
}
